// Model_39694087750057
// MI455X (gfx1250) — matrix operand order repaired and checked by reading, not compiled, not run on hardware
//
#include <hip/hip_runtime.h>

typedef __attribute__((ext_vector_type(16))) __bf16 v16bf;
typedef __attribute__((ext_vector_type(8)))  __bf16 v8bf;
typedef __attribute__((ext_vector_type(8)))  float  v8f;
typedef __attribute__((ext_vector_type(4)))  unsigned int u32x4;
typedef __attribute__((ext_vector_type(8)))  int i32x8;
typedef __attribute__((ext_vector_type(4)))  int i32x4;

#if defined(__AMDGCN__) && __has_builtin(__builtin_amdgcn_tensor_load_to_lds) && \
    __has_builtin(__builtin_amdgcn_s_wait_tensorcnt)
#define HAVE_TDM 1
#else
#define HAVE_TDM 0
#endif

// ---------- bf16 helpers (native converts -> v_cvt_*bf16* ops) ----------
static __device__ __forceinline__ unsigned short f32_to_bf16_bits(float f) {
  return __builtin_bit_cast(unsigned short, (__bf16)f);
}
static __device__ __forceinline__ __bf16 f32_to_bf16(float f) {
  return (__bf16)f;
}
static __device__ __forceinline__ float bf16_bits_to_f32(unsigned short h) {
  unsigned int u = ((unsigned int)h) << 16;
  return __builtin_bit_cast(float, u);
}

// ============================================================
// Weight pre-conversion: f32 -> bf16, 8 elements / thread.
// ============================================================
__global__ __launch_bounds__(256)
void convert_f32_bf16(const float* __restrict__ in,
                      unsigned short* __restrict__ out, int n8)
{
  const int i = blockIdx.x * 256 + threadIdx.x;
  if (i >= n8) return;
  const float4 a = ((const float4*)in)[2 * i];
  const float4 b = ((const float4*)in)[2 * i + 1];
  uint4 o;
  o.x = (unsigned)f32_to_bf16_bits(a.x) | ((unsigned)f32_to_bf16_bits(a.y) << 16);
  o.y = (unsigned)f32_to_bf16_bits(a.z) | ((unsigned)f32_to_bf16_bits(a.w) << 16);
  o.z = (unsigned)f32_to_bf16_bits(b.x) | ((unsigned)f32_to_bf16_bits(b.y) << 16);
  o.w = (unsigned)f32_to_bf16_bits(b.z) | ((unsigned)f32_to_bf16_bits(b.w) << 16);
  ((uint4*)out)[i] = o;
}

// ============================================================
// WMMA GEMM:  out[M,N] = op( A1@W1^T + b1 (+ A2@W2^T + b2) )
// A* : f32 row-major [M,K], converted to bf16 in-register (native v_cvt).
// W* : bf16 row-major [N,K], pre-converted. The block's 64xK W-strip is
//      staged into LDS once per pass -- by the Tensor Data Mover when the
//      builtin exists (D# per ISA 8.3/8.4, pad 4 DWORDs every 128 DWORDs
//      -> 264-elem LDS pitch), else by a cooperative copy. Fragments are
//      then read with ds_load_b128 (row pitch 528B == 4 banks: 16 lanes
//      cover 16 distinct 4-bank groups -> conflict-free).
// Block = 256 thr = 8 waves; each wave owns a 16x64 strip (4 WMMA N-tiles
// sharing one A fragment). Grid: (M/128, N/64). K == 256 here.
// ============================================================
template<bool DUAL, bool RELU, bool OUT_BF16>
__global__ __launch_bounds__(256)
void gemm_wmma_bf16(const float* __restrict__ A1, const unsigned short* __restrict__ W1,
                    const float* __restrict__ bias1,
                    const float* __restrict__ A2, const unsigned short* __restrict__ W2,
                    const float* __restrict__ bias2,
                    float* __restrict__ Cf, unsigned short* __restrict__ Cb,
                    int M, int N, int K)
{
  constexpr int PITCH = 264;                  // 256 + 8 bf16 pad (16B) per row
  __shared__ unsigned short smem[64 * PITCH]; // 33,792 B

  const int lane = threadIdx.x & 31;
  const int wave = threadIdx.x >> 5;
  const int m    = lane & 15;   // row-within-tile for A, col-within-tile for B/C
  const int hhf  = lane >> 4;   // half-wave select (ISA 16-bit fragment layout)
  const int row0 = blockIdx.x * 128 + wave * 16;
  const int col0 = blockIdx.y * 64;

  v8f acc[4];
  #pragma unroll
  for (int n = 0; n < 4; ++n) acc[n] = (v8f){0.f,0.f,0.f,0.f,0.f,0.f,0.f,0.f};

  const int npass = DUAL ? 2 : 1;
  for (int p = 0; p < npass; ++p) {
    const float* __restrict__ A = (DUAL && p) ? A2 : A1;
    const unsigned short* __restrict__ W = (DUAL && p) ? W2 : W1;

    __syncthreads();   // protect smem reuse between passes
#if HAVE_TDM
    if (wave == 0) {
      // ---- Tensor DMA: 64 x K tile of 2B elems, global -> LDS, padded ----
      const unsigned long long ga =
          (unsigned long long)(const void*)(W + (size_t)col0 * K);
      const unsigned lds_base = (unsigned)(unsigned long long)(const void*)&smem[0];
      u32x4 g0;
      g0[0] = 1u;                                   // count=1, is_restore=0
      g0[1] = lds_base;                             // lds_addr
      g0[2] = (unsigned)ga;                         // global_addr[31:0]
      g0[3] = (unsigned)((ga >> 32) & 0x01FFFFFFu)  // global_addr[56:32]
              | (2u << 30);                         // type=2 ("image")
      i32x8 g1;
      g1[0] = (int)((1u << 16)      // data_size = 2B
              | (1u << 20)          // pad_enable
              | (6u << 22)          // pad_interval: 128 DWORDs (=256 bf16 row)
              | (3u << 25));        // pad_amount: 4 DWORDs (=8 bf16 -> pitch 264)
      g1[1] = (int)(((unsigned)K & 0xFFFFu) << 16);            // tensor_dim0 lo16
      g1[2] = (int)((((unsigned)K >> 16) & 0xFFFFu)            // tensor_dim0 hi16
              | (((unsigned)N & 0xFFFFu) << 16));              // tensor_dim1 lo16
      g1[3] = (int)((((unsigned)N >> 16) & 0xFFFFu)            // tensor_dim1 hi16
              | (((unsigned)K & 0xFFFFu) << 16));              // tile_dim0 = K
      g1[4] = 64;                                              // tile_dim1 = 64
      g1[5] = (int)(unsigned)K;                                // dim0_stride lo32
      g1[6] = 0;
      g1[7] = 0;
      const i32x4 gz = (i32x4){0, 0, 0, 0};
#if __clang_major__ >= 23
      __builtin_amdgcn_tensor_load_to_lds(g0, g1, gz, gz,
                                          (i32x8){0,0,0,0,0,0,0,0}, 0);
#else
      __builtin_amdgcn_tensor_load_to_lds(g0, g1, gz, gz, 0);
#endif
      __builtin_amdgcn_s_wait_tensorcnt((short)0);
    }
#else
    // ---- fallback: cooperative staged copy (global_load + ds_store) ----
    #pragma unroll
    for (int i = 0; i < 8; ++i) {
      const int t  = threadIdx.x + i * 256;  // 0..2047 16B-chunks
      const int r  = t >> 5;                 // row 0..63
      const int c8 = (t & 31) * 8;           // elem offset 0..248
      const uint4 v = *(const uint4*)(W + (size_t)(col0 + r) * K + c8);
      *(uint4*)(&smem[r * PITCH + c8]) = v;
    }
#endif
    __syncthreads();

    const float* arow = A + (size_t)(row0 + m) * K;
    for (int kt = 0; kt < K; kt += 32) {
      // ---- A fragment 16x32 bf16: lane<16 holds K {kt+0..7, kt+16..23},
      //                             lane>=16 holds K {kt+8..15, kt+24..31}
      const int k0 = kt + hhf * 8;
      const float4 a0 = *(const float4*)(arow + k0);
      const float4 a1 = *(const float4*)(arow + k0 + 4);
      const float4 a2 = *(const float4*)(arow + k0 + 16);
      const float4 a3 = *(const float4*)(arow + k0 + 20);
      if (kt + 32 < K) __builtin_prefetch(arow + k0 + 64, 0, 3); // global_prefetch_b8

      v16bf a;
      a[0]=f32_to_bf16(a0.x);  a[1]=f32_to_bf16(a0.y);  a[2]=f32_to_bf16(a0.z);  a[3]=f32_to_bf16(a0.w);
      a[4]=f32_to_bf16(a1.x);  a[5]=f32_to_bf16(a1.y);  a[6]=f32_to_bf16(a1.z);  a[7]=f32_to_bf16(a1.w);
      a[8]=f32_to_bf16(a2.x);  a[9]=f32_to_bf16(a2.y);  a[10]=f32_to_bf16(a2.z); a[11]=f32_to_bf16(a2.w);
      a[12]=f32_to_bf16(a3.x); a[13]=f32_to_bf16(a3.y); a[14]=f32_to_bf16(a3.z); a[15]=f32_to_bf16(a3.w);

      #pragma unroll
      for (int n = 0; n < 4; ++n) {
        // ---- B fragment 32x16 bf16 from LDS: lane j=m holds
        //      K = kt+hhf*8..+7 and kt+16+hhf*8..+7 -> 2x ds_load_b128
        const unsigned short* wr = &smem[(n * 16 + m) * PITCH + kt + hhf * 8];
        const v8bf blo = *(const v8bf*)(wr);
        const v8bf bhi = *(const v8bf*)(wr + 16);
        const v16bf b = __builtin_shufflevector(blo, bhi,
            0,1,2,3,4,5,6,7,8,9,10,11,12,13,14,15);

        acc[n] = __builtin_amdgcn_wmma_f32_16x16x32_bf16(
            /*neg_a=*/false, a, /*neg_b=*/false, b,
            /*c_mod=*/(short)0, acc[n], /*reuse_a=*/false, /*reuse_b=*/false);
      }
    }
  }

  // ---- C/D layout: VGPR v, lanes 0-15 -> M=v, N=lane; lanes 16-31 -> M=v+8
  #pragma unroll
  for (int n = 0; n < 4; ++n) {
    const int colg = col0 + n * 16 + m;
    float bi = bias1[colg];
    if (DUAL) bi += bias2[colg];
    #pragma unroll
    for (int v = 0; v < 8; ++v) {
      const int rowg = row0 + v + hhf * 8;
      const size_t idx = (size_t)rowg * N + colg;
      float val = acc[n][v] + bi;
      if (RELU) val = val > 0.f ? val : 0.f;
      if (OUT_BF16) Cb[idx] = f32_to_bf16_bits(val);
      else          Cf[idx] = val;
    }
  }
}

// ============================================================
// Edge kernel: segment-max of (pooled[src[e]] * w[e]) into neigh[dst[e]].
// One wave per edge, 8 feats per lane (F=256). Messages are >= 0 (relu * w>=0),
// so f32 atomic-max == u32 atomic-max on bit patterns; zero-init reproduces
// the reference's where(isfinite, neigh, 0) for empty segments.
// ============================================================
__global__ __launch_bounds__(256)
void edge_segment_max(const unsigned short* __restrict__ pooled,
                      const float* __restrict__ w,
                      const int* __restrict__ src,
                      const int* __restrict__ dst,
                      unsigned int* __restrict__ neigh,
                      int E)
{
  const int gid  = blockIdx.x * 256 + threadIdx.x;
  const int e    = gid >> 5;
  const int lane = gid & 31;
  if (e >= E) return;
  const int   s  = src[e];
  const int   d  = dst[e];
  const float we = w[e];

  const unsigned short* prow = pooled + (size_t)s * 256 + lane * 8;
  const uint4 pk = *(const uint4*)prow;                 // 8 bf16 = 16B
  unsigned int* nrow = neigh + (size_t)d * 256 + lane * 8;

  const unsigned int words[4] = { pk.x, pk.y, pk.z, pk.w };
  #pragma unroll
  for (int i = 0; i < 4; ++i) {
    const float lo = bf16_bits_to_f32((unsigned short)(words[i] & 0xFFFFu)) * we;
    const float hv = bf16_bits_to_f32((unsigned short)(words[i] >> 16))     * we;
    atomicMax(nrow + 2 * i,     __builtin_bit_cast(unsigned int, lo));
    atomicMax(nrow + 2 * i + 1, __builtin_bit_cast(unsigned int, hv));
  }
}

// ============================================================
extern "C" void kernel_launch(void* const* d_in, const int* in_sizes, int n_in,
                              void* d_out, int out_size, void* d_ws, size_t ws_size,
                              hipStream_t stream) {
  (void)in_sizes; (void)n_in; (void)out_size; (void)ws_size;
  const float* x        = (const float*)d_in[0];
  const int*   src1     = (const int*)  d_in[1];
  const int*   dst1     = (const int*)  d_in[2];
  const float* w1       = (const float*)d_in[3];
  const int*   src2     = (const int*)  d_in[4];
  const int*   dst2     = (const int*)  d_in[5];
  const float* w2       = (const float*)d_in[6];
  const float* pool_w1  = (const float*)d_in[7];
  const float* pool_b1  = (const float*)d_in[8];
  const float* self_w1  = (const float*)d_in[9];
  const float* self_b1  = (const float*)d_in[10];
  const float* neigh_w1 = (const float*)d_in[11];
  const float* neigh_b1 = (const float*)d_in[12];
  const float* pool_w2  = (const float*)d_in[13];
  const float* pool_b2  = (const float*)d_in[14];
  const float* self_w2  = (const float*)d_in[15];
  const float* self_b2  = (const float*)d_in[16];
  const float* neigh_w2 = (const float*)d_in[17];
  const float* neigh_b2 = (const float*)d_in[18];

  constexpr int IN_F = 256, H1 = 256, H2 = 128;
  constexpr int NS1 = 262144, ND1 = 32768, E1 = 524288;
  constexpr int ND2 = 4096,   E2  = 65536;

  // ---- workspace layout (~223 MB) ----
  char* ws = (char*)d_ws;
  size_t off = 0;
  unsigned short* pooled1 = (unsigned short*)(ws + off); off += (size_t)NS1 * IN_F * 2;
  float*          neigh1  = (float*)         (ws + off); off += (size_t)ND1 * H1  * 4;
  float*          h       = (float*)         (ws + off); off += (size_t)ND1 * H1  * 4;
  unsigned short* pooled2 = (unsigned short*)(ws + off); off += (size_t)ND1 * H1  * 2;
  float*          neigh2  = (float*)         (ws + off); off += (size_t)ND2 * H1  * 4;
  unsigned short* pw1b    = (unsigned short*)(ws + off); off += (size_t)H1 * IN_F * 2;
  unsigned short* sw1b    = (unsigned short*)(ws + off); off += (size_t)H1 * IN_F * 2;
  unsigned short* nw1b    = (unsigned short*)(ws + off); off += (size_t)H1 * IN_F * 2;
  unsigned short* pw2b    = (unsigned short*)(ws + off); off += (size_t)H1 * H1   * 2;
  unsigned short* sw2b    = (unsigned short*)(ws + off); off += (size_t)H2 * H1   * 2;
  unsigned short* nw2b    = (unsigned short*)(ws + off); off += (size_t)H2 * H1   * 2;

  hipMemsetAsync(neigh1, 0, (size_t)ND1 * H1 * 4, stream);
  hipMemsetAsync(neigh2, 0, (size_t)ND2 * H1 * 4, stream);

  const dim3 blk(256);

  // ---- pre-convert the 6 weight matrices to bf16 (tiny) ----
  {
    const int n8_big = H1 * IN_F / 8;  // 8192
    const int n8_sm  = H2 * H1  / 8;   // 4096
    convert_f32_bf16<<<dim3((n8_big + 255) / 256), blk, 0, stream>>>(pool_w1,  pw1b, n8_big);
    convert_f32_bf16<<<dim3((n8_big + 255) / 256), blk, 0, stream>>>(self_w1,  sw1b, n8_big);
    convert_f32_bf16<<<dim3((n8_big + 255) / 256), blk, 0, stream>>>(neigh_w1, nw1b, n8_big);
    convert_f32_bf16<<<dim3((n8_big + 255) / 256), blk, 0, stream>>>(pool_w2,  pw2b, n8_big);
    convert_f32_bf16<<<dim3((n8_sm  + 255) / 256), blk, 0, stream>>>(self_w2,  sw2b, n8_sm);
    convert_f32_bf16<<<dim3((n8_sm  + 255) / 256), blk, 0, stream>>>(neigh_w2, nw2b, n8_sm);
  }

  // Layer 1: pooled1 = relu(x @ pool_w1^T + pool_b1)  [262144 x 256] -> bf16
  gemm_wmma_bf16<false, true, true><<<dim3(NS1/128, IN_F/64), blk, 0, stream>>>(
      x, pw1b, pool_b1, nullptr, nullptr, nullptr, nullptr, pooled1, NS1, IN_F, IN_F);

  // neigh1 = segment_max(pooled1[src1] * w1, dst1)
  edge_segment_max<<<dim3(E1 * 32 / 256), blk, 0, stream>>>(
      pooled1, w1, src1, dst1, (unsigned int*)neigh1, E1);

  // h = relu(x[:ND1]@self_w1^T + neigh1@neigh_w1^T + self_b1 + neigh_b1) -> f32
  gemm_wmma_bf16<true, true, false><<<dim3(ND1/128, H1/64), blk, 0, stream>>>(
      x, sw1b, self_b1, neigh1, nw1b, neigh_b1, h, nullptr, ND1, H1, IN_F);

  // Layer 2: pooled2 = relu(h @ pool_w2^T + pool_b2)  [32768 x 256] -> bf16
  gemm_wmma_bf16<false, true, true><<<dim3(ND1/128, H1/64), blk, 0, stream>>>(
      h, pw2b, pool_b2, nullptr, nullptr, nullptr, nullptr, pooled2, ND1, H1, H1);

  // neigh2 = segment_max(pooled2[src2] * w2, dst2)
  edge_segment_max<<<dim3(E2 * 32 / 256), blk, 0, stream>>>(
      pooled2, w2, src2, dst2, (unsigned int*)neigh2, E2);

  // out = h[:ND2]@self_w2^T + neigh2@neigh_w2^T + self_b2 + neigh_b2 -> f32
  float* out = (float*)d_out;
  gemm_wmma_bf16<true, false, false><<<dim3(ND2/128, H2/64), blk, 0, stream>>>(
      h, sw2b, self_b2, neigh2, nw2b, neigh_b2, out, nullptr, ND2, H2, H1);
}
